// PointNet2Stage_12429635355325
// MI455X (gfx1250) — compile-verified, matrix operand order checked by reading, not run on hardware
//
#include <hip/hip_runtime.h>

typedef __attribute__((ext_vector_type(16))) __bf16 v16bf;
typedef __attribute__((ext_vector_type(8)))  __bf16 v8bf;
typedef __attribute__((ext_vector_type(8)))  float  v8f;

// ---------------------------------------------------------------------------
// Pack a row-major f32 weight [K, Nout] into bf16 WMMA B-fragment order:
// tiles of (32K x 16N), 512 bf16 per tile, lane-major, 16 elements per lane
// in VGPR order (reg r, pair p):  K_local = (r&4?16:0) + (lane&16?8:0) + 2*(r&3) + p,
// N = tile_n*16 + (lane&15).  GEMM then loads each B fragment with 2x b128.
// ---------------------------------------------------------------------------
__global__ __launch_bounds__(256) void pack_weight_kernel(const float* __restrict__ W,
                                                          __bf16* __restrict__ P,
                                                          int K, int Nout) {
  int i = blockIdx.x * 256 + threadIdx.x;
  if (i >= K * Nout) return;
  int pair = i & 1;
  int r    = (i >> 1) & 7;
  int lane = (i >> 4) & 31;
  int tile = i >> 9;
  int nct  = Nout >> 4;
  int kt   = tile / nct;
  int nt   = tile - kt * nct;
  int n = (nt << 4) + (lane & 15);
  int klocal = ((r & 4) ? 16 : 0) + ((lane & 16) ? 8 : 0) + ((r & 3) << 1) + pair;
  int k = (kt << 5) + klocal;
  P[i] = (__bf16)W[k * Nout + n];
}

// ---------------------------------------------------------------------------
// Stage 1: pf1 = relu(x @ W1 + b1), K=6 (too skinny for WMMA; 0.3% of FLOPs).
// ---------------------------------------------------------------------------
__global__ __launch_bounds__(256) void point_stage1_kernel(
    const float* __restrict__ x, const float* __restrict__ W1,
    const float* __restrict__ b1, __bf16* __restrict__ pf1, int N)
{
  __shared__ float sW[6 * 64];
  __shared__ float sb[64];
  for (int i = threadIdx.x; i < 6 * 64; i += 256) sW[i] = W1[i];
  if (threadIdx.x < 64) sb[threadIdx.x] = b1[threadIdx.x];
  __syncthreads();
  int p = blockIdx.x * 256 + threadIdx.x;
  if (p >= N) return;
  float xi[6];
#pragma unroll
  for (int c = 0; c < 6; ++c) xi[c] = x[(size_t)p * 6 + c];
  __bf16* op = pf1 + (size_t)p * 64;
#pragma unroll 8
  for (int j = 0; j < 64; ++j) {
    float a = sb[j];
#pragma unroll
    for (int c = 0; c < 6; ++c) a += xi[c] * sW[c * 64 + j];
    op[j] = (__bf16)(a > 0.f ? a : 0.f);
  }
}

__global__ __launch_bounds__(256) void f32_to_bf16_kernel(const float* __restrict__ x,
                                                          __bf16* __restrict__ y, int n) {
  int i = blockIdx.x * 256 + threadIdx.x;
  if (i < n) y[i] = (__bf16)x[i];
}

// ---------------------------------------------------------------------------
// Fused GEMM:  D = relu(A[M x KT*32] @ Wp + bias)  via v_wmma_f32_16x16x32_bf16.
// Fully compile-time specialized: k-loop unrolls, concat source selection is
// resolved per unrolled k-step, epilogue modes are if-constexpr.
// Block = 8 waves; each wave computes a 32x64 output tile (2 row-tiles x 4
// col-tiles, 8 x v8f accumulators); B fragments reused across row tiles.
// FLAGS: 1 = A row is concat(A2[idx[row]][0:128], A[row][0:128]) (KT==8)
//        2 = store bf16 rows to outBf
//        4 = scatter atomic-max (uint-bits trick, values >= 0) into voxMax[idx[m]]
//        8 = store f32 rows to outF
// ---------------------------------------------------------------------------
template <int FLAGS, int KT, int NOUT>
__global__ __launch_bounds__(256) void wmma_mlp_kernel(
    const __bf16* __restrict__ A,
    const __bf16* __restrict__ A2,
    const int*    __restrict__ idx,
    const __bf16* __restrict__ Wp,
    const float*  __restrict__ bias,
    __bf16* __restrict__ outBf,
    float*  __restrict__ voxMax,
    float*  __restrict__ outF,
    int M)
{
  constexpr int K = KT << 5;
  constexpr int nWaveCols = NOUT >> 6;               // waves per row group
  constexpr int rowsPerBlock = (8 / nWaveCols) << 5; // 128 (NOUT=128) or 64 (NOUT=256)
  constexpr int nct = NOUT >> 4;

  const int lane = threadIdx.x & 31;
  const int wave = threadIdx.x >> 5;
  const int waveRow = wave / nWaveCols;
  const int waveCol = wave - waveRow * nWaveCols;
  const int m0 = blockIdx.x * rowsPerBlock + (waveRow << 5); // 32 rows per wave
  const int n0 = waveCol << 6;

  // A-fragment addressing: row = m0 + g*16 + lane%16, K-half by lane/16.
  // baseLo serves k-steps in [0,128) (gathered half for concat), baseHi the
  // rest; baseHi is pre-biased by -128 elements so base + k works directly.
  const int khalf = (lane >> 4) << 3;
  const __bf16* baseLo[2];
  const __bf16* baseHi[2];
#pragma unroll
  for (int g = 0; g < 2; ++g) {
    int rowA = m0 + (g << 4) + (lane & 15);
    int rowC = rowA < M ? rowA : (M - 1);
    if constexpr (FLAGS & 1) {
      baseLo[g] = A2 + (size_t)idx[rowC] * 128;
      baseHi[g] = A + (size_t)rowC * 128 - 128;
    } else {
      baseLo[g] = A + (size_t)rowC * K;
      baseHi[g] = baseLo[g];
    }
  }

  v8f acc[8] = {};

#pragma unroll
  for (int kt = 0; kt < KT; ++kt) {
    const int kA = (kt << 5) + khalf;
    const int kB = kA + 16;
    constexpr bool useHi0 = false;  // placeholder to keep structure clear
    (void)useHi0;

    // ---- load both A fragments (source half known at compile time) ----
    v16bf afrag[2];
#pragma unroll
    for (int g = 0; g < 2; ++g) {
      const __bf16* base = ((FLAGS & 1) && (kt * 32 >= 128)) ? baseHi[g] : baseLo[g];
      v8bf lo = *(const v8bf*)(base + kA);
      v8bf hi = *(const v8bf*)(base + kB);
#pragma unroll
      for (int e = 0; e < 8; ++e) { afrag[g][e] = lo[e]; afrag[g][8 + e] = hi[e]; }
    }

    // ---- load all four B fragments (distinct regs -> one load clause) ----
    const __bf16* wtile = Wp + (((size_t)(kt * nct + (n0 >> 4))) << 9) + lane * 16;
    v16bf bfrag[4];
#pragma unroll
    for (int t = 0; t < 4; ++t) {
      const __bf16* bp = wtile + ((size_t)t << 9);
      v8bf blo = *(const v8bf*)bp;
      v8bf bhi = *(const v8bf*)(bp + 8);
#pragma unroll
      for (int e = 0; e < 8; ++e) { bfrag[t][e] = blo[e]; bfrag[t][8 + e] = bhi[e]; }
    }

    // ---- 8 WMMAs: each B fragment reused for both row tiles ----
#pragma unroll
    for (int t = 0; t < 4; ++t) {
#pragma unroll
      for (int g = 0; g < 2; ++g) {
        acc[t * 2 + g] = __builtin_amdgcn_wmma_f32_16x16x32_bf16(
            false, afrag[g], false, bfrag[t], (short)0, acc[t * 2 + g], false, false);
      }
    }
  }

  // Epilogue. C/D layout: VGPR r, lane l -> M = base + r + 8*(l>=16), N = n0 + l%16.
  int vIdx[2][8];
  if constexpr (FLAGS & 4) {
#pragma unroll
    for (int g = 0; g < 2; ++g)
#pragma unroll
      for (int r = 0; r < 8; ++r) {
        int m = m0 + (g << 4) + r + ((lane >> 4) << 3);
        vIdx[g][r] = (m < M) ? idx[m] : 0;
      }
  }

#pragma unroll
  for (int t = 0; t < 4; ++t) {
    const int n = n0 + (t << 4) + (lane & 15);
    const float bv = bias[n];
#pragma unroll
    for (int g = 0; g < 2; ++g) {
#pragma unroll
      for (int r = 0; r < 8; ++r) {
        const int m = m0 + (g << 4) + r + ((lane >> 4) << 3);
        if (m >= M) continue;
        float v = acc[t * 2 + g][r] + bv;
        v = v > 0.f ? v : 0.f;
        if constexpr (FLAGS & 2) outBf[(size_t)m * NOUT + n] = (__bf16)v;
        if constexpr (FLAGS & 4)
          atomicMax((unsigned int*)(voxMax + (size_t)vIdx[g][r] * NOUT + n),
                    __float_as_uint(v));   // valid: v >= 0
        if constexpr (FLAGS & 8) outF[(size_t)m * NOUT + n] = v;
      }
    }
  }
}

// ---------------------------------------------------------------------------
extern "C" void kernel_launch(void* const* d_in, const int* in_sizes, int n_in,
                              void* d_out, int out_size, void* d_ws, size_t ws_size,
                              hipStream_t stream) {
  (void)in_sizes; (void)n_in; (void)out_size; (void)ws_size;
  const float* inp = (const float*)d_in[0];
  const int*   idx = (const int*)  d_in[1];
  const float* W1  = (const float*)d_in[2];
  const float* b1  = (const float*)d_in[3];
  const float* W2  = (const float*)d_in[4];
  const float* b2  = (const float*)d_in[5];
  const float* Wv1 = (const float*)d_in[6];
  const float* bv1 = (const float*)d_in[7];
  const float* W3  = (const float*)d_in[8];
  const float* b3  = (const float*)d_in[9];
  const float* W4  = (const float*)d_in[10];
  const float* b4  = (const float*)d_in[11];
  const float* Wv2 = (const float*)d_in[12];
  const float* bv2 = (const float*)d_in[13];
  float* out = (float*)d_out;

  const int N = 500000, V = 65536;

  char* ws = (char*)d_ws;
  size_t off = 0;
  auto wsAlloc = [&](size_t bytes) -> char* {
    char* p = ws + off;
    off = (off + bytes + 255) & ~(size_t)255;
    return p;
  };

  __bf16* W2p   = (__bf16*)wsAlloc((size_t)64  * 128 * 2);
  __bf16* Wv1p  = (__bf16*)wsAlloc((size_t)128 * 128 * 2);
  __bf16* W3p   = (__bf16*)wsAlloc((size_t)256 * 256 * 2);
  __bf16* W4p   = (__bf16*)wsAlloc((size_t)256 * 256 * 2);
  __bf16* Wv2p  = (__bf16*)wsAlloc((size_t)256 * 256 * 2);
  __bf16* pf1   = (__bf16*)wsAlloc((size_t)N * 256 * 2); // union: pf1 [N,64] then pf4 [N,256]
  __bf16* pf4   = pf1;
  __bf16* pf2   = (__bf16*)wsAlloc((size_t)N * 128 * 2);
  float*  vox1  = (float*) wsAlloc((size_t)V * 128 * 4);
  __bf16* vox1b = (__bf16*)wsAlloc((size_t)V * 128 * 2);
  __bf16* voxr  = (__bf16*)wsAlloc((size_t)V * 128 * 2);
  float*  vox2  = (float*) wsAlloc((size_t)V * 256 * 4);
  __bf16* vox2b = (__bf16*)wsAlloc((size_t)V * 256 * 2);

  // atomic-max targets must start at 0 every call (matches segment_max0 semantics)
  hipMemsetAsync(vox1, 0, (size_t)V * 128 * 4, stream);
  hipMemsetAsync(vox2, 0, (size_t)V * 256 * 4, stream);

  pack_weight_kernel<<<(64 * 128 + 255) / 256, 256, 0, stream>>>(W2,  W2p,  64, 128);
  pack_weight_kernel<<<(128 * 128 + 255) / 256, 256, 0, stream>>>(Wv1, Wv1p, 128, 128);
  pack_weight_kernel<<<(256 * 256 + 255) / 256, 256, 0, stream>>>(W3,  W3p,  256, 256);
  pack_weight_kernel<<<(256 * 256 + 255) / 256, 256, 0, stream>>>(W4,  W4p,  256, 256);
  pack_weight_kernel<<<(256 * 256 + 255) / 256, 256, 0, stream>>>(Wv2, Wv2p, 256, 256);

  // pf1 = relu(x @ W1 + b1)                              [N,64] bf16
  point_stage1_kernel<<<(N + 255) / 256, 256, 0, stream>>>(inp, W1, b1, pf1, N);

  // pf2 = relu(pf1 @ W2 + b2); scatter-max -> vox1       [N,128] + [V,128]
  wmma_mlp_kernel<2 | 4, 2, 128><<<(N + 127) / 128, 256, 0, stream>>>(
      pf1, nullptr, idx, W2p, b2, pf2, vox1, nullptr, N);
  f32_to_bf16_kernel<<<(V * 128 + 255) / 256, 256, 0, stream>>>(vox1, vox1b, V * 128);

  // voxr = relu(vox1 @ Wv1 + bv1)                        [V,128]
  wmma_mlp_kernel<2, 4, 128><<<(V + 127) / 128, 256, 0, stream>>>(
      vox1b, nullptr, nullptr, Wv1p, bv1, voxr, nullptr, nullptr, V);

  // pf4 = relu(concat(voxr[idx], pf2) @ W3 + b3)         [N,256]
  wmma_mlp_kernel<1 | 2, 8, 256><<<(N + 63) / 64, 256, 0, stream>>>(
      pf2, voxr, idx, W3p, b3, pf4, nullptr, nullptr, N);

  // pf5 = relu(pf4 @ W4 + b4) -> scatter-max only -> vox2 (no row store)
  wmma_mlp_kernel<4, 8, 256><<<(N + 63) / 64, 256, 0, stream>>>(
      pf4, nullptr, idx, W4p, b4, nullptr, vox2, nullptr, N);
  f32_to_bf16_kernel<<<(V * 256 + 255) / 256, 256, 0, stream>>>(vox2, vox2b, V * 256);

  // out = relu(vox2 @ Wv2 + bv2)                         [V,256] f32 -> d_out
  wmma_mlp_kernel<8, 8, 256><<<(V + 63) / 64, 256, 0, stream>>>(
      vox2b, nullptr, nullptr, Wv2p, bv2, nullptr, nullptr, out, V);
}
